// SSIMLoss_89060441850534
// MI455X (gfx1250) — matrix operand order repaired and checked by reading, not compiled, not run on hardware
//
#include <hip/hip_runtime.h>
#include <hip/hip_bf16.h>

typedef __attribute__((ext_vector_type(16))) _Float16 v16h;
typedef __attribute__((ext_vector_type(8)))  _Float16 v8h;
typedef __attribute__((ext_vector_type(8)))  float    v8f;

#define IMG   512
#define OUTD  502         // 512 - 11 + 1
#define NTOT  16128256.0f // 64 * 502 * 502

// Normalized 11-tap Gaussian, sigma = 1.5 (precomputed in double, matches ref;
// operands are rounded to f16 for WMMA anyway, so f32 constants are exact enough).
__device__ __constant__ const float GW[11] = {
    0.00102838f, 0.00759880f, 0.03600077f, 0.10936069f, 0.21300553f,
    0.26601172f,
    0.21300553f, 0.10936069f, 0.03600077f, 0.00759880f, 0.00102838f
};

// ---------------------------------------------------------------------------
// Main kernel: one 256-thread WG computes a 64x64 output tile of one plane.
// H pass: WMMA(A = input rows f16, B = Gaussian band) -> Ht (transposed f16 LDS)
// V pass: WMMA(A = Gaussian band, B = Ht columns)     -> 5 conv fragments in regs
// then SSIM map + deterministic reduction to partial[blockIdx.x].
// ---------------------------------------------------------------------------
__global__ __launch_bounds__(256)
void ssim_main(const float* __restrict__ X, const float* __restrict__ Y,
               float* __restrict__ partial)
{
    // Ht[s][col][row]: 5 streams, 64 output cols, 80 rows (stride 80 halves =
    // 160B keeps every v8h access 16B aligned).
    __shared__ _Float16 Ht[5 * 64 * 80];
    __shared__ float    wpart[8];

    const int tid  = threadIdx.x;
    // wave index is wave-uniform: force it scalar so WMMA loops stay in SALU
    // control flow with EXEC all-ones (ISA requirement for WMMA).
    const int wave = __builtin_amdgcn_readfirstlane(tid >> 5);
    const int lane = tid & 31;
    const int hi   = lane >> 4;   // half-wave select in WMMA layouts
    const int ln   = lane & 15;   // M (A) / N (B,C,D) index

    const int wg    = blockIdx.x;
    const int plane = wg >> 6;          // n*4 + c
    const int t8    = wg & 63;
    const int r0    = (t8 >> 3) * 64;   // tile origin in output coords
    const int c0    = (t8 & 7) * 64;
    const size_t pbase = (size_t)plane * (IMG * IMG);

    // Constant B operand for the H pass: B[K][c] = g[K - c] band.
    // 16-bit B 32x16 layout: element t of lane -> K = (t&7) + 8*((t>>3)*2 + hi), N = ln.
    v16h Bh;
#pragma unroll
    for (int t = 0; t < 16; ++t) {
        int d = (t & 7) + 8 * ((t >> 3) * 2 + hi) - ln;
        Bh[t] = (_Float16)((d >= 0 && d <= 10) ? GW[d] : 0.0f);
    }

    // ---------------- H pass: 20 positions (i=0..4 row blocks, j=0..3 col
    // blocks), 5 streams each -> 100 WMMAs per WG, strided across 8 waves.
    for (int m = wave; m < 20; m += 8) {
        const int i = m >> 2, j = m & 3;
        const int rowbase = i * 16, colbase = j * 16;

        // A gather straight from global (edge-clamped: clamped values only feed
        // masked outputs >= 502, and stay finite so 0*x in WMMA is safe).
        const int arow = r0 + rowbase + ln;
        const int row  = (arow < IMG - 1) ? arow : (IMG - 1);
        const size_t rbg = pbase + (size_t)row * IMG;

        float xv[16], yv[16];
#pragma unroll
        for (int t = 0; t < 16; ++t) {
            int K   = (t & 7) + 8 * ((t >> 3) * 2 + hi);   // 16-bit A 16x32 layout
            int col = c0 + colbase + K;
            col = (col < IMG - 1) ? col : (IMG - 1);
            xv[t] = X[rbg + col];
            yv[t] = Y[rbg + col];
        }
        v16h Ax, Ay, Axx, Ayy, Axy;
#pragma unroll
        for (int t = 0; t < 16; ++t) {
            Ax[t]  = (_Float16)xv[t];
            Ay[t]  = (_Float16)yv[t];
            Axx[t] = (_Float16)(xv[t] * xv[t]);
            Ayy[t] = (_Float16)(yv[t] * yv[t]);
            Axy[t] = (_Float16)(xv[t] * yv[t]);
        }

        v8f c0f = {}, c1f = {}, c2f = {}, c3f = {}, c4f = {};
        c0f = __builtin_amdgcn_wmma_f32_16x16x32_f16(false, Ax,  false, Bh, (short)0, c0f, false, false);
        c1f = __builtin_amdgcn_wmma_f32_16x16x32_f16(false, Ay,  false, Bh, (short)0, c1f, false, false);
        c2f = __builtin_amdgcn_wmma_f32_16x16x32_f16(false, Axx, false, Bh, (short)0, c2f, false, false);
        c3f = __builtin_amdgcn_wmma_f32_16x16x32_f16(false, Ayy, false, Bh, (short)0, c3f, false, false);
        c4f = __builtin_amdgcn_wmma_f32_16x16x32_f16(false, Axy, false, Bh, (short)0, c4f, false, false);

        // Store transposed: D element p of lane -> row M = p + 8*hi, col N = ln.
        // Ht[s][colbase+ln][rowbase + 8*hi + p] => one aligned 16B store/stream.
        const int hoff = (colbase + ln) * 80 + rowbase + 8 * hi;
        v8f dc[5] = { c0f, c1f, c2f, c3f, c4f };
#pragma unroll
        for (int s = 0; s < 5; ++s) {
            v8h hv;
#pragma unroll
            for (int p = 0; p < 8; ++p) hv[p] = (_Float16)dc[s][p];
            *(v8h*)(&Ht[s * 5120 + hoff]) = hv;
        }
    }
    __syncthreads();

    // Constant A operand for the V pass: A[r][K] = g[K - r] band.
    v16h Av;
#pragma unroll
    for (int t = 0; t < 16; ++t) {
        int d = t + 16 * hi - ln;
        Av[t] = (_Float16)((d >= 0 && d <= 10) ? GW[d] : 0.0f);
    }

    // ---------------- V pass + SSIM map: 16 output tiles, 5 WMMAs each,
    // all 5 streams of a tile in the same wave -> fragments stay in registers.
    const float C1 = 1e-4f, C2 = 9e-4f;
    float accl = 0.0f;

    for (int t2 = wave; t2 < 16; t2 += 8) {
        const int i = t2 >> 2, j = t2 & 3;
        const int rbs = i * 16, cbs = j * 16;

        v8f acc[5];
#pragma unroll
        for (int s = 0; s < 5; ++s) {
            // B element t -> K = t + 16*hi: Ht row rbs + 16*hi + t, col cbs+ln.
            const _Float16* Hp = &Ht[s * 5120 + (cbs + ln) * 80 + rbs + 16 * hi];
            v8h b0 = *(const v8h*)(Hp);
            v8h b1 = *(const v8h*)(Hp + 8);
            v16h B;
#pragma unroll
            for (int e = 0; e < 8; ++e) { B[e] = b0[e]; B[e + 8] = b1[e]; }
            v8f c = {};
            acc[s] = __builtin_amdgcn_wmma_f32_16x16x32_f16(false, Av, false, B, (short)0, c, false, false);
        }

        const int gcc = c0 + cbs + ln;
#pragma unroll
        for (int p = 0; p < 8; ++p) {
            const int grr = r0 + rbs + 8 * hi + p;
            if (grr < OUTD && gcc < OUTD) {
                float muX = acc[0][p], muY = acc[1][p];
                float exx = acc[2][p], eyy = acc[3][p], exy = acc[4][p];
                float mxx = muX * muX, myy = muY * muY, mxy = muX * muY;
                float sxx = exx - mxx, syy = eyy - myy, sxy = exy - mxy;
                float num = (2.0f * mxy + C1) * (2.0f * sxy + C2);
                float den = (mxx + myy + C1) * (sxx + syy + C2);
                accl += num / den;
            }
        }
    }

    // Deterministic reduction: shuffle tree within wave, fixed order across waves.
#pragma unroll
    for (int off = 16; off > 0; off >>= 1) accl += __shfl_xor(accl, off, 32);
    if (lane == 0) wpart[wave] = accl;
    __syncthreads();
    if (tid == 0) {
        float s = 0.0f;
#pragma unroll
        for (int w = 0; w < 8; ++w) s += wpart[w];
        partial[wg] = s;
    }
}

// ---------------------------------------------------------------------------
// Final reduction of 4096 per-WG partials -> d_out[0] = 1 - mean(ssim_map)
// ---------------------------------------------------------------------------
__global__ __launch_bounds__(256)
void ssim_reduce(const float* __restrict__ partial, float* __restrict__ out)
{
    __shared__ float sm[256];
    const int t = threadIdx.x;
    float s = 0.0f;
#pragma unroll
    for (int k = 0; k < 16; ++k) s += partial[t + 256 * k];
    sm[t] = s;
    __syncthreads();
    for (int st = 128; st > 0; st >>= 1) {
        if (t < st) sm[t] += sm[t + st];
        __syncthreads();
    }
    if (t == 0) out[0] = 1.0f - sm[0] / NTOT;
}

extern "C" void kernel_launch(void* const* d_in, const int* in_sizes, int n_in,
                              void* d_out, int out_size, void* d_ws, size_t ws_size,
                              hipStream_t stream)
{
    (void)in_sizes; (void)n_in; (void)out_size; (void)ws_size;
    const float* X = (const float*)d_in[0];
    const float* Y = (const float*)d_in[1];
    float* ws  = (float*)d_ws;    // 4096 per-WG partial sums (16 KB)
    float* out = (float*)d_out;   // scalar f32

    ssim_main<<<dim3(4096), dim3(256), 0, stream>>>(X, Y, ws);
    ssim_reduce<<<dim3(1), dim3(256), 0, stream>>>(ws, out);
}
